// Attention_65180423685674
// MI455X (gfx1250) — matrix operand order repaired and checked by reading, not compiled, not run on hardware
//
#include <hip/hip_runtime.h>
#include <hip/hip_bf16.h>

// Shapes fixed by the reference setup.
#define B_SZ   8
#define NN     16384      // H*W = 128*128
#define CC     64         // DIM = HEAD_DIM (1 head)
#define NK     256        // (128/8)*(128/8) reduced tokens
#define QSCALE 0.125f     // 64^-0.5

typedef __attribute__((ext_vector_type(16))) __bf16         v16bf;
typedef __attribute__((ext_vector_type(8)))  float          v8f;
typedef __attribute__((ext_vector_type(16))) unsigned short v16u;
typedef __attribute__((ext_vector_type(8)))  unsigned short v8u;

static __device__ __forceinline__ unsigned short f2bf(float f) {
  union { float f; unsigned u; } c; c.f = f;
  unsigned u = c.u + 0x7FFFu + ((c.u >> 16) & 1u);   // round-to-nearest-even
  return (unsigned short)(u >> 16);
}

static __device__ __forceinline__ v8f wmma_bf16(v16u a, v16u b, v8f c) {
  return __builtin_amdgcn_wmma_f32_16x16x32_bf16(
      false, __builtin_bit_cast(v16bf, a),
      false, __builtin_bit_cast(v16bf, b),
      (short)0, c, false, false);
}

static __device__ __forceinline__ v16u cat8(v8u lo, v8u hi) {
  return __builtin_shufflevector(lo, hi, 0,1,2,3,4,5,6,7,8,9,10,11,12,13,14,15);
}

// ---------------------------------------------------------------------------
// K1: transpose + bf16-convert Wq and Wproj (B-matrix wants [out][in]).
// ---------------------------------------------------------------------------
__global__ void __launch_bounds__(256)
cvt_weights_kernel(const float* __restrict__ Wq, const float* __restrict__ Wproj,
                   unsigned short* __restrict__ WqT, unsigned short* __restrict__ WprojT) {
  int t = blockIdx.x * blockDim.x + threadIdx.x;
  if (t < 64 * 64) {
    int o = t >> 6, d = t & 63;
    WqT[o * 64 + d]    = f2bf(Wq[d * 64 + o]);
    WprojT[o * 64 + d] = f2bf(Wproj[d * 64 + o]);
  }
}

// ---------------------------------------------------------------------------
// K2: spatial-reduction conv (8x8/s8) + bias + LayerNorm + KV projection.
// One 256-thread block per (batch, patch). Writes k row-major [b][j][d]
// (already B^T for the scores GEMM) and v transposed [b][d][j] (B^T for ctx).
// ---------------------------------------------------------------------------
__global__ void __launch_bounds__(256)
sr_ln_kv_kernel(const float* __restrict__ x, const float* __restrict__ sr_w,
                const float* __restrict__ sr_b, const float* __restrict__ ln_g,
                const float* __restrict__ ln_b, const float* __restrict__ Wkv,
                unsigned short* __restrict__ kb, unsigned short* __restrict__ vb) {
  __shared__ float part[4][64];
  __shared__ float y[64];
  __shared__ float xn[64];
  const int bid = blockIdx.x;            // b*256 + patch
  const int b = bid >> 8, j = bid & 255;
  const int py = j >> 4, px = j & 15;
  const int t = threadIdx.x;
  const int o = t & 63, chunk = t >> 6;  // 4 K-chunks (2 kernel rows each)

  float acc = 0.f;
  for (int kyi = 0; kyi < 2; ++kyi) {
    const int ky = chunk * 2 + kyi;
    for (int kx = 0; kx < 8; ++kx) {
      const float* xp = x + ((long)b * NN + (long)(py * 8 + ky) * 128 + (px * 8 + kx)) * CC;
      const float* wp = sr_w + (long)o * 4096 + ky * 8 + kx;   // [o][c][ky][kx], c stride 64
#pragma unroll 8
      for (int c = 0; c < 64; ++c) acc = fmaf(xp[c], wp[c * 64], acc);
    }
  }
  part[chunk][o] = acc;
  __syncthreads();
  if (t < 64) y[t] = part[0][t] + part[1][t] + part[2][t] + part[3][t] + sr_b[t];
  __syncthreads();

  float mu = 0.f;
  for (int c = 0; c < 64; ++c) mu += y[c];
  mu *= (1.f / 64.f);
  float var = 0.f;
  for (int c = 0; c < 64; ++c) { float d = y[c] - mu; var = fmaf(d, d, var); }
  var *= (1.f / 64.f);
  const float rs = rsqrtf(var + 1e-5f);
  if (t < 64) xn[t] = (y[t] - mu) * rs * ln_g[t] + ln_b[t];
  __syncthreads();

  if (t < 128) {
    float dot = 0.f;
    for (int c = 0; c < 64; ++c) dot = fmaf(xn[c], Wkv[c * 128 + t], dot);
    if (t < 64) kb[((long)(b * NK) + j) * CC + t]      = f2bf(dot);       // k[b][j][d]
    else        vb[((long)b * CC + (t - 64)) * NK + j] = f2bf(dot);       // v^T[b][d][j]
  }
}

// ---------------------------------------------------------------------------
// K3: q = (x @ Wq) * SCALE  -> bf16.  One wave per 16-row tile, WMMA bf16.
// ---------------------------------------------------------------------------
__global__ void __launch_bounds__(128)
qproj_kernel(const float* __restrict__ x, const unsigned short* __restrict__ WqT,
             unsigned short* __restrict__ qb) {
  const int wave = threadIdx.x >> 5, lane = threadIdx.x & 31;
  const int lo = lane & 15, hi = lane >> 4, s0 = hi ? 8 : 0;
  const long tile = (long)blockIdx.x * 4 + wave;     // over B*N/16 rows
  const long row0 = tile << 4;
  const v8f vzero = {0.f, 0.f, 0.f, 0.f, 0.f, 0.f, 0.f, 0.f};
  v8f acc[4] = {vzero, vzero, vzero, vzero};

#pragma unroll
  for (int ks = 0; ks < 2; ++ks) {                   // K = 64 -> 2 steps of 32
    const float* xr = x + (row0 + lo) * CC + ks * 32;
    v16u av;
#pragma unroll
    for (int i = 0; i < 8; ++i) av[i] = f2bf(xr[hi * 16 + i]);
#pragma unroll
    for (int i = 0; i < 8; ++i) av[8 + i] = f2bf(xr[hi * 16 + 8 + i]);
#pragma unroll
    for (int nt = 0; nt < 4; ++nt) {
      v16u bv = *(const v16u*)(WqT + (nt * 16 + lo) * CC + ks * 32 + hi * 16);
      acc[nt] = wmma_bf16(av, bv, acc[nt]);
    }
  }
#pragma unroll
  for (int nt = 0; nt < 4; ++nt)
#pragma unroll
    for (int r = 0; r < 8; ++r)
      qb[(row0 + r + 8 * hi) * CC + nt * 16 + lo] = f2bf(acc[nt][r] * QSCALE);
}

// ---------------------------------------------------------------------------
// K4: fused attention. One wave per 16-row q tile, 4 waves/block (a block
// never straddles a batch: 256 blocks per batch). The batch's K matrix is
// staged into LDS once per block with GLOBAL_LOAD_ASYNC_TO_LDS_B128
// (ASYNCcnt, drained with s_wait_asynccnt), padded to a conflict-free
// stride; the same LDS is then overlaid by the per-wave P / ctx tiles.
// scores -> softmax (in-reg + shfl) -> ctx (P@v) -> fused output projection.
// ---------------------------------------------------------------------------
#define KLDS_STRIDE 72                    // 64 + 8 ushorts = 144 B rows
#define P_STRIDE    264                   // 256 + 8 ushorts
#define P_TILE      (16 * P_STRIDE)       // per-wave ushorts

__global__ void __launch_bounds__(128)
attn_kernel(const unsigned short* __restrict__ qb, const unsigned short* __restrict__ kb,
            const unsigned short* __restrict__ vb, const unsigned short* __restrict__ WprojT,
            const float* __restrict__ bproj, float* __restrict__ out) {
  // 256 rows * 72 ushorts = 36864 B. Reused later as 4 x P_TILE (33792 B).
  __shared__ __align__(32) unsigned short sh[NK * KLDS_STRIDE];
  const int tid  = threadIdx.x;
  const int wave = tid >> 5, lane = tid & 31;
  const int lo = lane & 15, hi = lane >> 4, s0 = hi ? 8 : 0;
  const long tile = (long)blockIdx.x * 4 + wave;
  const int  b    = (int)(blockIdx.x >> 8);          // 256 blocks per batch
  const long row0 = tile << 4;
  const v8f vzero = {0.f, 0.f, 0.f, 0.f, 0.f, 0.f, 0.f, 0.f};

  __builtin_prefetch(vb + (long)b * CC * NK, 0, 2);  // global_prefetch_b8
  __builtin_prefetch(WprojT, 0, 2);

  // ---- Stage K[b] (256 x 64 bf16) into LDS via async DMA, padded rows. ----
  {
    const unsigned short* kg = kb + (long)b * NK * CC;
#pragma unroll
    for (int i = 0; i < 16; ++i) {                   // 2048 16B chunks / 128 thr
      const int c   = tid + i * 128;
      const int row = c >> 3, seg = c & 7;
      unsigned           ldst = (unsigned)(size_t)(sh + row * KLDS_STRIDE + seg * 8);
      unsigned long long ga   = (unsigned long long)(size_t)(kg + row * 64 + seg * 8);
      asm volatile("global_load_async_to_lds_b128 %0, %1, off"
                   :: "v"(ldst), "v"(ga) : "memory");
    }
    asm volatile("s_wait_asynccnt 0x0" ::: "memory");
  }
  __syncthreads();

  // q A-fragments (scale pre-folded), K = 64 -> 2 steps.
  v16u aq[2];
#pragma unroll
  for (int ks = 0; ks < 2; ++ks) {
    const unsigned short* p = qb + (row0 + lo) * CC + ks * 32;
    aq[ks] = cat8(*(const v8u*)(p + hi * 16), *(const v8u*)(p + hi * 16 + 8));
  }

  // scores: 16 column tiles x 2 K-steps = 32 WMMAs; B-frags from LDS K.
  v8f sc[16];
#pragma unroll
  for (int jt = 0; jt < 16; ++jt) {
    const unsigned short* kp = sh + (jt * 16 + lo) * KLDS_STRIDE + hi * 16;
    v8f a = vzero;
    a = wmma_bf16(aq[0], *(const v16u*)(kp),      a);
    a = wmma_bf16(aq[1], *(const v16u*)(kp + 32), a);
    sc[jt] = a;
  }

  // softmax over 256 cols per row: reg-reduce over 16 tiles + shfl over 16 lanes.
  float inv[8];
#pragma unroll
  for (int r = 0; r < 8; ++r) {
    float m = -3.0e38f;
#pragma unroll
    for (int jt = 0; jt < 16; ++jt) m = fmaxf(m, sc[jt][r]);
    m = fmaxf(m, __shfl_xor(m, 1, 32));
    m = fmaxf(m, __shfl_xor(m, 2, 32));
    m = fmaxf(m, __shfl_xor(m, 4, 32));
    m = fmaxf(m, __shfl_xor(m, 8, 32));
    float s = 0.f;
#pragma unroll
    for (int jt = 0; jt < 16; ++jt) {
      float e = __expf(sc[jt][r] - m);
      sc[jt][r] = e;
      s += e;
    }
    s += __shfl_xor(s, 1, 32);
    s += __shfl_xor(s, 2, 32);
    s += __shfl_xor(s, 4, 32);
    s += __shfl_xor(s, 8, 32);
    inv[r] = __builtin_amdgcn_rcpf(s);
  }

  // All waves done reading the K staging region before it is overlaid by P.
  __syncthreads();

  // P (bf16) -> per-wave LDS tile (stride 264) for re-read in A layout.
  unsigned short* P = sh + wave * P_TILE;
#pragma unroll
  for (int jt = 0; jt < 16; ++jt)
#pragma unroll
    for (int r = 0; r < 8; ++r)
      P[(r + 8 * hi) * P_STRIDE + jt * 16 + lo] = f2bf(sc[jt][r] * inv[r]);
  __syncthreads();

  // ctx = P @ v : K = 256 -> 8 steps, N = 64 -> 4 tiles (32 WMMAs).
  v8f cacc[4] = {vzero, vzero, vzero, vzero};
#pragma unroll
  for (int ks = 0; ks < 8; ++ks) {
    const unsigned short* pr = P + lo * P_STRIDE + ks * 32;
    v16u ap = cat8(*(const v8u*)(pr + hi * 16), *(const v8u*)(pr + hi * 16 + 8));
#pragma unroll
    for (int nt = 0; nt < 4; ++nt) {
      const unsigned short* vp = vb + ((long)b * CC + nt * 16 + lo) * NK + ks * 32 + hi * 16;
      cacc[nt] = wmma_bf16(ap, *(const v16u*)vp, cacc[nt]);
    }
  }
  __syncthreads();

  // ctx -> LDS (stride 72), then fused 64x64 output projection (8 WMMAs).
#pragma unroll
  for (int nt = 0; nt < 4; ++nt)
#pragma unroll
    for (int r = 0; r < 8; ++r)
      P[(r + 8 * hi) * 72 + nt * 16 + lo] = f2bf(cacc[nt][r]);
  __syncthreads();

  v8f pacc[4] = {vzero, vzero, vzero, vzero};
#pragma unroll
  for (int ks = 0; ks < 2; ++ks) {
    const unsigned short* pr = P + lo * 72 + ks * 32;
    v16u ap = cat8(*(const v8u*)(pr + hi * 16), *(const v8u*)(pr + hi * 16 + 8));
#pragma unroll
    for (int nt = 0; nt < 4; ++nt) {
      v16u bv = *(const v16u*)(WprojT + (nt * 16 + lo) * CC + ks * 32 + hi * 16);
      pacc[nt] = wmma_bf16(ap, bv, pacc[nt]);
    }
  }
#pragma unroll
  for (int nt = 0; nt < 4; ++nt) {
    const float bbias = bproj[nt * 16 + lo];
#pragma unroll
    for (int r = 0; r < 8; ++r)
      out[(row0 + r + 8 * hi) * CC + nt * 16 + lo] = pacc[nt][r] + bbias;
  }
}

// ---------------------------------------------------------------------------
extern "C" void kernel_launch(void* const* d_in, const int* in_sizes, int n_in,
                              void* d_out, int out_size, void* d_ws, size_t ws_size,
                              hipStream_t stream) {
  const float* x     = (const float*)d_in[0];
  const float* Wq    = (const float*)d_in[1];
  const float* Wkv   = (const float*)d_in[2];
  const float* Wproj = (const float*)d_in[3];
  const float* bproj = (const float*)d_in[4];
  const float* sr_w  = (const float*)d_in[5];
  const float* sr_b  = (const float*)d_in[6];
  const float* ln_g  = (const float*)d_in[7];
  const float* ln_b  = (const float*)d_in[8];
  // d_in[9], d_in[10] = H, W (fixed 128 in setup; not needed on device)

  // Workspace layout (bf16 buffers).
  unsigned short* qb     = (unsigned short*)d_ws;                 // [B*N][64]   16 MB
  unsigned short* kb     = qb + (size_t)B_SZ * NN * CC;           // [B][256][64]
  unsigned short* vb     = kb + (size_t)B_SZ * NK * CC;           // [B][64][256] (v^T)
  unsigned short* WqT    = vb + (size_t)B_SZ * CC * NK;           // [64][64]
  unsigned short* WprojT = WqT + 64 * 64;                         // [64][64]

  cvt_weights_kernel<<<16, 256, 0, stream>>>(Wq, Wproj, WqT, WprojT);
  sr_ln_kv_kernel<<<B_SZ * NK, 256, 0, stream>>>(x, sr_w, sr_b, ln_g, ln_b, Wkv, kb, vb);
  qproj_kernel<<<(B_SZ * NN / 16) / 4, 128, 0, stream>>>(x, WqT, qb);
  attn_kernel<<<(B_SZ * NN / 16) / 4, 128, 0, stream>>>(qb, kb, vb, WprojT, bproj,
                                                        (float*)d_out);
  (void)in_sizes; (void)n_in; (void)out_size; (void)ws_size;
}
